// IpaScore_2894807957719
// MI455X (gfx1250) — matrix operand order repaired and checked by reading, not compiled, not run on hardware
//
#include <hip/hip_runtime.h>
#include <math.h>

typedef __attribute__((ext_vector_type(16))) _Float16 v16h;
typedef __attribute__((ext_vector_type(8)))  _Float16 v8h;
typedef __attribute__((ext_vector_type(4)))  _Float16 v4h;
typedef __attribute__((ext_vector_type(8)))  float    v8f;

#define NRES 384
#define NN   (384*384)

// ---------------------------------------------------------------------------
// Generic WMMA GEMM: C = act((alpha * A@B(+T) + bias + P[i]+Q[j]) * rowscale + add)
// A: (M,K) lda ; B: (K,N) ldb  (or (N,K) if BTRANS) ; batched via blockIdx.z.
// 256 threads = 8 waves; 64x64 tile; each wave owns a 32x16 strip (2 WMMAs).
// LDS double-buffered: stage tile k+1 while computing tile k.
// Requires K % 4 == 0 and lda % 4 == 0 (true for every call here).
// ---------------------------------------------------------------------------
struct GemmP {
    const float* A; long lda, sA;
    const float* B; long ldb, sB;
    const float* bias;
    const float* add; long ldadd, sAdd;
    const float* rowscale;            // per-row scale on (alpha*prod+bias) before add
    const float* P; const float* Q;   // EDGEADD: + P[row/NJ][n] + Q[row%NJ][n]
    int NJ, ldPQ;
    float* C; long ldc, sC;
    int M, N, K; float alpha;
};

#define LDSP 48   // padded row stride (halves); 96B, 16B-aligned for b128

template<int BTRANS, int RELU, int EDGEADD>
__global__ __launch_bounds__(256) void k_gemm(GemmP p) {
    __shared__ _Float16 As[2][64][LDSP];   // [buf][m][k]
    __shared__ _Float16 Bs[2][64][LDSP];   // [buf][n][k]  (transposed stage)
    const int bz = blockIdx.z;
    const float* A = p.A + (long)bz * p.sA;
    const float* B = p.B + (long)bz * p.sB;
    float*       C = p.C + (long)bz * p.sC;
    const float* add = p.add ? p.add + (long)bz * p.sAdd : nullptr;
    const int M = p.M, N = p.N, K = p.K;
    const long lda = p.lda, ldb = p.ldb;

    const int m0 = blockIdx.y * 64;
    const int n0 = blockIdx.x * 64;
    const int tid = threadIdx.x, lane = tid & 31, wave = tid >> 5;
    const int mw = (wave >> 2) << 5;   // 0/32
    const int nw = (wave & 3) << 4;    // 0/16/32/48
    const bool bvec = ((ldb & 3) == 0) && ((N & 3) == 0);
    v8f acc0 = {0.f,0.f,0.f,0.f,0.f,0.f,0.f,0.f};
    v8f acc1 = {0.f,0.f,0.f,0.f,0.f,0.f,0.f,0.f};

    auto stage = [&](int buf, int kb) {
        // ---- A tile (64x32): float4 loads, clamped addresses, branch-free ----
        #pragma unroll
        for (int it = 0; it < 2; ++it) {
            int f  = tid + it*256;          // 0..511 float4 chunks
            int mm = f >> 3;                // 0..63
            int k4 = (f & 7) << 2;          // 0..28
            int gm = m0 + mm, gk = kb + k4;
            int gmc = gm < M ? gm : M - 1;
            int gkc = (gk + 4 <= K) ? gk : K - 4;
            float4 v4 = *(const float4*)&A[(long)gmc * lda + gkc];
            float ok = (gm < M && gk < K) ? 1.f : 0.f;
            v4h h;
            h[0] = (_Float16)(v4.x * ok); h[1] = (_Float16)(v4.y * ok);
            h[2] = (_Float16)(v4.z * ok); h[3] = (_Float16)(v4.w * ok);
            *(v4h*)&As[buf][mm][k4] = h;
        }
        // ---- B tile (32x64 / 64x32) into Bs[n][k] ----
        if (BTRANS) {           // B is (N,K): float4 along K -> contiguous store
            #pragma unroll
            for (int it = 0; it < 2; ++it) {
                int f  = tid + it*256;
                int nn = f >> 3;
                int k4 = (f & 7) << 2;
                int gn = n0 + nn, gk = kb + k4;
                int gnc = gn < N ? gn : N - 1;
                int gkc = (gk + 4 <= K) ? gk : K - 4;
                float4 v4 = *(const float4*)&B[(long)gnc * ldb + gkc];
                float ok = (gn < N && gk < K) ? 1.f : 0.f;
                v4h h;
                h[0] = (_Float16)(v4.x * ok); h[1] = (_Float16)(v4.y * ok);
                h[2] = (_Float16)(v4.z * ok); h[3] = (_Float16)(v4.w * ok);
                *(v4h*)&Bs[buf][nn][k4] = h;
            }
        } else if (bvec) {      // B is (K,N): float4 along N, transpose on store
            #pragma unroll
            for (int it = 0; it < 2; ++it) {
                int f  = tid + it*256;
                int kk = f >> 4;                 // 0..31
                int n4 = (f & 15) << 2;          // 0..60
                int gk = kb + kk, gn = n0 + n4;
                int gkc = gk < K ? gk : K - 1;
                int gnc = (gn + 4 <= N) ? gn : N - 4;
                float4 v4 = *(const float4*)&B[(long)gkc * ldb + gnc];
                float ok = (gk < K && gn < N) ? 1.f : 0.f;
                Bs[buf][n4+0][kk] = (_Float16)(v4.x * ok);
                Bs[buf][n4+1][kk] = (_Float16)(v4.y * ok);
                Bs[buf][n4+2][kk] = (_Float16)(v4.z * ok);
                Bs[buf][n4+3][kk] = (_Float16)(v4.w * ok);
            }
        } else {                // tiny-N fallback: clamped scalar loads, branch-free
            for (int idx = tid; idx < 64*32; idx += 256) {
                int kk = idx >> 6, nn = idx & 63;
                int gk = kb + kk, gn = n0 + nn;
                int gkc = gk < K ? gk : K - 1;
                int gnc = gn < N ? gn : N - 1;
                float v = B[(long)gkc * ldb + gnc];
                v = (gk < K && gn < N) ? v : 0.f;
                Bs[buf][nn][kk] = (_Float16)v;
            }
        }
    };

    const int nb = (K + 31) >> 5;
    stage(0, 0);
    for (int ib = 0; ib < nb; ++ib) {
        __syncthreads();                      // staged[ib] visible; buf[ib^1] free
        int cur = ib & 1;
        if (ib + 1 < nb) stage(cur ^ 1, (ib + 1) << 5);
        if (ib + 2 < nb) {                    // prefetch two tiles ahead
            int gm = m0 + (tid & 63);
            if (gm < M) __builtin_prefetch(&A[(long)gm * lda + ((ib + 2) << 5)], 0, 1);
        }
        // ---- fragments (all b128 LDS loads) + 2 WMMAs sharing B ----
        const int half = lane >> 4;
        const int mr0 = mw + (lane & 15);
        const int mr1 = mr0 + 16;
        const int nc  = nw + (lane & 15);
        v8h a00 = *(const v8h*)&As[cur][mr0][8*half];
        v8h a01 = *(const v8h*)&As[cur][mr0][16 + 8*half];
        v8h a10 = *(const v8h*)&As[cur][mr1][8*half];
        v8h a11 = *(const v8h*)&As[cur][mr1][16 + 8*half];
        v8h b0  = *(const v8h*)&Bs[cur][nc][8*half];
        v8h b1  = *(const v8h*)&Bs[cur][nc][16 + 8*half];
        v16h af0 = __builtin_shufflevector(a00, a01, 0,1,2,3,4,5,6,7,8,9,10,11,12,13,14,15);
        v16h af1 = __builtin_shufflevector(a10, a11, 0,1,2,3,4,5,6,7,8,9,10,11,12,13,14,15);
        v16h bf  = __builtin_shufflevector(b0,  b1,  0,1,2,3,4,5,6,7,8,9,10,11,12,13,14,15);
        acc0 = __builtin_amdgcn_wmma_f32_16x16x32_f16(false, af0, false, bf,
                                                      (short)0, acc0, false, false);
        acc1 = __builtin_amdgcn_wmma_f32_16x16x32_f16(false, af1, false, bf,
                                                      (short)0, acc1, false, false);
    }
    // ---- epilogue: element r -> row = base + r + 8*(lane/16), col = lane%16 ----
    const int ncg = n0 + nw + (lane & 15);
    if (ncg < N) {
        #pragma unroll
        for (int sub = 0; sub < 2; ++sub) {
            int rb = m0 + mw + (sub << 4) + ((lane >> 4) << 3);
            #pragma unroll
            for (int r = 0; r < 8; ++r) {
                int gm = rb + r;
                if (gm < M) {
                    float v = (sub ? acc1[r] : acc0[r]) * p.alpha;
                    if (p.bias) v += p.bias[ncg];
                    if (EDGEADD) v += p.P[(long)(gm / p.NJ) * p.ldPQ + ncg]
                                   + p.Q[(long)(gm % p.NJ) * p.ldPQ + ncg];
                    if (p.rowscale) v *= p.rowscale[gm];
                    if (add) v += add[(long)gm * p.ldadd + ncg];
                    if (RELU) v = v > 0.f ? v : 0.f;
                    C[(long)gm * p.ldc + ncg] = v;
                }
            }
        }
    }
}

static void gemm(hipStream_t st, int BT, int RL, int EA, const GemmP& p, int batch) {
    dim3 g((p.N + 63) / 64, (p.M + 63) / 64, batch), b(256);
    if (BT)            hipLaunchKernelGGL(HIP_KERNEL_NAME(k_gemm<1,0,0>), g, b, 0, st, p);
    else if (EA && RL) hipLaunchKernelGGL(HIP_KERNEL_NAME(k_gemm<0,1,1>), g, b, 0, st, p);
    else if (EA)       hipLaunchKernelGGL(HIP_KERNEL_NAME(k_gemm<0,0,1>), g, b, 0, st, p);
    else if (RL)       hipLaunchKernelGGL(HIP_KERNEL_NAME(k_gemm<0,1,0>), g, b, 0, st, p);
    else               hipLaunchKernelGGL(HIP_KERNEL_NAME(k_gemm<0,0,0>), g, b, 0, st, p);
}
static GemmP GP(const float* A, long lda, long sA, const float* B, long ldb, long sB,
                const float* bias, float* C, long ldc, long sC,
                int M, int N, int K, float alpha) {
    GemmP p{}; p.A=A; p.lda=lda; p.sA=sA; p.B=B; p.ldb=ldb; p.sB=sB; p.bias=bias;
    p.add=nullptr; p.ldadd=0; p.sAdd=0; p.rowscale=nullptr; p.P=nullptr; p.Q=nullptr;
    p.NJ=1; p.ldPQ=0; p.C=C; p.ldc=ldc; p.sC=sC; p.M=M; p.N=N; p.K=K; p.alpha=alpha;
    return p;
}

// ---------------------------------------------------------------------------
// Block reductions (blockDim == 128)
// ---------------------------------------------------------------------------
__device__ inline float red_sum(float v, float* red) {
    int t = threadIdx.x; red[t] = v; __syncthreads();
    for (int s = 64; s > 0; s >>= 1) { if (t < s) red[t] += red[t + s]; __syncthreads(); }
    float r = red[0]; __syncthreads(); return r;
}
__device__ inline float red_max(float v, float* red) {
    int t = threadIdx.x; red[t] = v; __syncthreads();
    for (int s = 64; s > 0; s >>= 1) { if (t < s) red[t] = fmaxf(red[t], red[t + s]); __syncthreads(); }
    float r = red[0]; __syncthreads(); return r;
}

// ---------------------------------------------------------------------------
// Init: mask node embed, rigids -> R,t, diffuse mask
// ---------------------------------------------------------------------------
__global__ void k_init(const float* node_in, const float* rigids, const float* res_mask,
                       const float* fixed_mask, float* s, float* init_ne,
                       float* R, float* t, float* dmask) {
    int n = blockIdx.x, tid = threadIdx.x;
    float m = res_mask[n];
    for (int d = tid; d < 256; d += blockDim.x) {
        float v = node_in[n*256 + d] * m;
        s[n*256 + d] = v; init_ne[n*256 + d] = v;
    }
    if (tid == 0) {
        const float* r = rigids + n*7;
        float w=r[0], x=r[1], y=r[2], z=r[3];
        float inv = rsqrtf(w*w + x*x + y*y + z*z);
        w*=inv; x*=inv; y*=inv; z*=inv;
        float* Rn = R + n*9;
        Rn[0]=1-2*(y*y+z*z); Rn[1]=2*(x*y-w*z); Rn[2]=2*(x*z+w*y);
        Rn[3]=2*(x*y+w*z);   Rn[4]=1-2*(x*x+z*z); Rn[5]=2*(y*z-w*x);
        Rn[6]=2*(x*z-w*y);   Rn[7]=2*(y*z+w*x);   Rn[8]=1-2*(x*x+y*y);
        t[n*3+0]=r[4]*0.1f; t[n*3+1]=r[5]*0.1f; t[n*3+2]=r[6]*0.1f;
        dmask[n] = (1.f - fixed_mask[n]) * m;
    }
}

// split_points + rigid apply: raw (N,192) comp-major -> qpt (N, H*PQ=64, 3)
__global__ void k_points_q(const float* raw, const float* R, const float* t, float* qpt) {
    int n = blockIdx.x, p = threadIdx.x;   // 64 threads
    const float* Rn = R + n*9; const float* tn = t + n*3;
    float vx = raw[n*192 + p], vy = raw[n*192 + 64 + p], vz = raw[n*192 + 128 + p];
    float* o = qpt + n*192 + p*3;
    o[0] = Rn[0]*vx + Rn[1]*vy + Rn[2]*vz + tn[0];
    o[1] = Rn[3]*vx + Rn[4]*vy + Rn[5]*vz + tn[1];
    o[2] = Rn[6]*vx + Rn[7]*vy + Rn[8]*vz + tn[2];
}
// raw (N,480) -> kpt (N,H,8,3), vpt (N,H,12,3)
__global__ void k_points_kv(const float* raw, const float* R, const float* t,
                            float* kpt, float* vpt) {
    int n = blockIdx.x, p = threadIdx.x;   // 160 threads, p = h*20+q
    const float* Rn = R + n*9; const float* tn = t + n*3;
    float vx = raw[n*480 + p], vy = raw[n*480 + 160 + p], vz = raw[n*480 + 320 + p];
    float gx = Rn[0]*vx + Rn[1]*vy + Rn[2]*vz + tn[0];
    float gy = Rn[3]*vx + Rn[4]*vy + Rn[5]*vz + tn[1];
    float gz = Rn[6]*vx + Rn[7]*vy + Rn[8]*vz + tn[2];
    int h = p / 20, q = p % 20;
    float* o = (q < 8) ? (kpt + n*192 + (h*8 + q)*3) : (vpt + n*288 + (h*12 + q - 8)*3);
    o[0]=gx; o[1]=gy; o[2]=gz;
}

// IPA logits (+bias,+point term,+mask) and row softmax, in place on a (H,N,N)
__global__ __launch_bounds__(128) void k_ipa_att(float* a, const float* biasz,
        const float* qpt, const float* kpt, const float* head_w, const float* mask) {
    int i = blockIdx.x, h = blockIdx.y, tid = threadIdx.x;
    __shared__ float qp[24];
    __shared__ float buf[NRES];
    __shared__ float red[128];
    if (tid < 24) qp[tid] = qpt[i*192 + h*24 + tid];
    __syncthreads();
    float hwv = head_w[h];
    float sp = (hwv > 20.f) ? hwv : log1pf(__expf(hwv));
    float hw = sp * 0.09622504486493763f;       // sqrt(1/108)
    float mi = mask[i];
    float* row = a + (long)h * NN + (long)i * NRES;
    float lm = -1e30f;
    for (int j = tid; j < NRES; j += 128) {
        const float* kp = kpt + j*192 + h*24;
        float pt = 0.f;
        #pragma unroll
        for (int p = 0; p < 8; ++p) {
            float dx = qp[p*3+0]-kp[p*3+0];
            float dy = qp[p*3+1]-kp[p*3+1];
            float dz = qp[p*3+2]-kp[p*3+2];
            pt += dx*dx + dy*dy + dz*dz;
        }
        float v = row[j] + 0.57735026918962576f * biasz[((long)i*NRES + j)*8 + h]
                  - 0.5f*hw*pt + 100000.0f*(mi*mask[j] - 1.0f);
        buf[j] = v; lm = fmaxf(lm, v);
    }
    __syncthreads();
    float mx = red_max(lm, red);
    float ls = 0.f;
    for (int j = tid; j < NRES; j += 128) { float e = __expf(buf[j]-mx); buf[j]=e; ls += e; }
    float sm = red_sum(ls, red);
    float inv = 1.f / sm;
    for (int j = tid; j < NRES; j += 128) row[j] = buf[j] * inv;
}

// transformer attention softmax with -1e9*(1-mask)
__global__ __launch_bounds__(128) void k_softmax_tfmr(float* att, const float* mask) {
    int i = blockIdx.x, h = blockIdx.y, tid = threadIdx.x;
    __shared__ float buf[NRES];
    __shared__ float red[128];
    float* row = att + (long)h * NN + (long)i * NRES;
    float lm = -1e30f;
    for (int j = tid; j < NRES; j += 128) {
        float v = row[j] - 1e9f * (1.f - mask[j]);
        buf[j] = v; lm = fmaxf(lm, v);
    }
    __syncthreads();
    float mx = red_max(lm, red);
    float ls = 0.f;
    for (int j = tid; j < NRES; j += 128) { float e = __expf(buf[j]-mx); buf[j]=e; ls += e; }
    float sm = red_sum(ls, red);
    float inv = 1.f / sm;
    for (int j = tid; j < NRES; j += 128) row[j] = buf[j] * inv;
}

// layernorm over D, optional row scale m1[row/NJ]*m2[row%NJ]; in-place safe
__global__ __launch_bounds__(128) void k_layernorm(const float* x, long ldx, float* out, long ldo,
        const float* g, const float* b, int D, const float* m1, const float* m2, int NJ) {
    long row = blockIdx.x;
    const float* xr = x + row * ldx;
    float* orow = out + row * ldo;
    __shared__ float red[128];
    int tid = threadIdx.x;
    float s = 0.f;
    for (int d = tid; d < D; d += 128) s += xr[d];
    float mean = red_sum(s, red) / (float)D;
    float v = 0.f;
    for (int d = tid; d < D; d += 128) { float dd = xr[d]-mean; v += dd*dd; }
    float var = red_sum(v, red) / (float)D;
    float inv = rsqrtf(var + 1e-5f);
    float scale = 1.f;
    if (m1) scale *= m1[row / NJ];
    if (m2) scale *= m2[row % NJ];
    for (int d = tid; d < D; d += 128)
        orow[d] = ((xr[d]-mean)*inv*g[d] + b[d]) * scale;
}

// o_pt -> local frame, fill cat[2048..2431]
__global__ void k_optlocal(const float* opt, const float* R, const float* t, float* cat) {
    int n = blockIdx.x, hp = threadIdx.x;   // 96 threads
    const float* Rn = R + n*9; const float* tn = t + n*3;
    const float* o = opt + n*288 + hp*3;
    float vx = o[0]-tn[0], vy = o[1]-tn[1], vz = o[2]-tn[2];
    float lx = Rn[0]*vx + Rn[3]*vy + Rn[6]*vz;   // R^T
    float ly = Rn[1]*vx + Rn[4]*vy + Rn[7]*vz;
    float lz = Rn[2]*vx + Rn[5]*vy + Rn[8]*vz;
    float* c = cat + (long)n*3456;
    c[2048 + hp] = lx; c[2048 + 96 + hp] = ly; c[2048 + 192 + hp] = lz;
    c[2336 + hp] = sqrtf(lx*lx + ly*ly + lz*lz + 1e-8f);
}

__global__ void k_copycols(const float* s, float* x) {   // s(384,256) -> x(384,320)[:,0:256]
    int n = blockIdx.x;
    for (int d = threadIdx.x; d < 256; d += blockDim.x) x[n*320 + d] = s[n*256 + d];
}
__global__ void k_maskrows(const float* in, const float* m, float* out, int D) {
    int r = blockIdx.x; float mm = m[r];
    for (int d = threadIdx.x; d < D; d += blockDim.x) out[(long)r*D + d] = in[(long)r*D + d]*mm;
}
__global__ void k_copyn(const float* a, float* b, int n) {
    int i = blockIdx.x*blockDim.x + threadIdx.x; if (i < n) b[i] = a[i];
}

__global__ void k_compose(const float* upd, const float* dmask, float* R, float* t) {
    int n = blockIdx.x*blockDim.x + threadIdx.x;
    if (n >= NRES) return;
    if (dmask[n] <= 0.f) return;
    const float* u = upd + n*6;
    float qx=u[0], qy=u[1], qz=u[2];
    float inv = rsqrtf(1.f + qx*qx + qy*qy + qz*qz);
    float w=inv, x=qx*inv, y=qy*inv, z=qz*inv;
    float U[9];
    U[0]=1-2*(y*y+z*z); U[1]=2*(x*y-w*z); U[2]=2*(x*z+w*y);
    U[3]=2*(x*y+w*z);   U[4]=1-2*(x*x+z*z); U[5]=2*(y*z-w*x);
    U[6]=2*(x*z-w*y);   U[7]=2*(y*z+w*x);   U[8]=1-2*(x*x+y*y);
    float* Rn = R + n*9;
    float Ro[9];
    #pragma unroll
    for (int k = 0; k < 9; ++k) Ro[k] = Rn[k];
    #pragma unroll
    for (int i2 = 0; i2 < 3; ++i2)
        #pragma unroll
        for (int j2 = 0; j2 < 3; ++j2) {
            float acc = 0.f;
            #pragma unroll
            for (int k2 = 0; k2 < 3; ++k2) acc += Ro[i2*3+k2]*U[k2*3+j2];
            Rn[i2*3+j2] = acc;
        }
    float tx=u[3], ty=u[4], tz=u[5];
    float* tn = t + n*3;
    float nx = Ro[0]*tx + Ro[1]*ty + Ro[2]*tz + tn[0];
    float ny = Ro[3]*tx + Ro[4]*ty + Ro[5]*tz + tn[1];
    float nz = Ro[6]*tx + Ro[7]*ty + Ro[8]*tz + tn[2];
    tn[0]=nx; tn[1]=ny; tn[2]=nz;
}

__global__ void k_out_tail(const float* un, const float* t, const float* R, float* out) {
    int n = blockIdx.x*blockDim.x + threadIdx.x;
    if (n >= NRES) return;
    float a = un[n*2], b = un[n*2+1];
    float inv = rsqrtf(fmaxf(a*a + b*b, 1e-8f));
    out[98304 + n*2]     = a*inv;
    out[98304 + n*2 + 1] = b*inv;
    #pragma unroll
    for (int k = 0; k < 3; ++k) out[99072 + n*3 + k] = t[n*3+k] * 10.0f;
    #pragma unroll
    for (int k = 0; k < 9; ++k) out[100224 + n*9 + k] = R[n*9+k];
}

// ---------------------------------------------------------------------------
// Host orchestration
// ---------------------------------------------------------------------------
struct Lin { const float* b; const float* w; };
struct LnP { const float* b; const float* g; };

extern "C" void kernel_launch(void* const* d_in, const int* in_sizes, int n_in,
                              void* d_out, int out_size, void* d_ws, size_t ws_size,
                              hipStream_t stream) {
    (void)in_sizes; (void)n_in; (void)out_size; (void)ws_size;
    hipStream_t st = stream;
    const float* node_in = (const float*)d_in[0];
    const float* edge_in = (const float*)d_in[1];
    const float* rigids  = (const float*)d_in[2];
    const float* mask    = (const float*)d_in[3];
    const float* fixedm  = (const float*)d_in[4];

    int ci = 5;
    auto NF = [&]() -> const float* { return (const float*)d_in[ci++]; };
    auto RL = [&]() { Lin l; l.b = NF(); l.w = NF(); return l; };
    auto RN = [&]() { LnP l; l.b = NF(); l.g = NF(); return l; };

    // workspace bump allocator
    char* wp = (char*)d_ws;
    auto alloc = [&](size_t nf) -> float* { float* p = (float*)wp; wp += nf * sizeof(float); return p; };
    float* Rb   = alloc(NRES*9);
    float* tb   = alloc(NRES*3);
    float* dmsk = alloc(NRES);
    float* s    = alloc(NRES*256);
    float* ine  = alloc(NRES*256);
    float* tmp1 = alloc(NRES*256);
    float* tmp2 = alloc(NRES*256);
    float* qb   = alloc(NRES*2048);
    float* kvb  = alloc(NRES*4096);
    float* qraw = alloc(NRES*192);
    float* kvraw= alloc(NRES*480);
    float* qpt  = alloc(NRES*192);
    float* kpt  = alloc(NRES*192);
    float* vpt  = alloc(NRES*288);
    float* bzb  = alloc((size_t)NN*8);
    float* ab   = alloc((size_t)NN*8);
    float* cat  = alloc(NRES*3456);
    float* optb = alloc(NRES*288);
    float* xb   = alloc(NRES*320);
    float* t320 = alloc(NRES*320);
    float* ffb  = alloc(NRES*320);
    float* qkvb = alloc(NRES*960);
    float* attb = alloc((size_t)NN*4);
    float* ob   = alloc(NRES*320);
    float* neb  = alloc(NRES*128);
    float* p1b  = alloc(NRES*384);
    float* q1b  = alloc(NRES*384);
    float* pfb  = alloc(NRES*128);
    float* qfb  = alloc(NRES*128);
    float* x1b  = alloc((size_t)48*384*384);
    float* x2b  = alloc((size_t)48*384*384);
    float* zA   = alloc((size_t)NN*128);
    float* zB   = alloc((size_t)NN*128);
    float* updb = alloc(NRES*6);
    float* unb  = alloc(NRES*2);

    k_init<<<NRES, 128, 0, st>>>(node_in, rigids, mask, fixedm, s, ine, Rb, tb, dmsk);

    const float* zcur = edge_in;
    bool flip = false;

    for (int b = 0; b < 4; ++b) {
        bool has_edge = (b < 3);
        // ---- read params in pytree-leaf order ----
        Lin bb = RL();
        Lin e_final{}, e_init{}, e_tr0{}, e_tr1{}; LnP e_ln{};
        if (has_edge) { e_final = RL(); e_init = RL(); e_ln = RN(); e_tr0 = RL(); e_tr1 = RL(); }
        Lin ipa_bias = RL();
        const float* head_w = NF();
        Lin kv = RL(); Lin kv_pts = RL(); Lin out = RL(); Lin q = RL(); Lin q_pts = RL();
        LnP ipa_ln = RN();
        Lin l1 = RL(); Lin l2 = RL(); Lin l3 = RL(); LnP nt_ln = RN();
        Lin post = RL(); Lin skip = RL();
        Lin tf_ff1[2], tf_ff2[2], tf_o[2], tf_qkv[2]; LnP tf_ln1[2], tf_ln2[2];
        for (int l = 0; l < 2; ++l) {
            tf_ff1[l]=RL(); tf_ff2[l]=RL(); tf_ln1[l]=RN(); tf_ln2[l]=RN();
            tf_o[l]=RL(); tf_qkv[l]=RL();
        }

        // ---- IPA ----
        gemm(st,0,0,0, GP(s,256,0,  q.w,2048,0,  q.b,  qb,2048,0,  NRES,2048,256, 1.f), 1);
        gemm(st,0,0,0, GP(s,256,0,  kv.w,4096,0, kv.b, kvb,4096,0, NRES,4096,256, 1.f), 1);
        gemm(st,0,0,0, GP(s,256,0,  q_pts.w,192,0,  q_pts.b,  qraw,192,0,  NRES,192,256, 1.f), 1);
        gemm(st,0,0,0, GP(s,256,0,  kv_pts.w,480,0, kv_pts.b, kvraw,480,0, NRES,480,256, 1.f), 1);
        gemm(st,0,0,0, GP(zcur,128,0, ipa_bias.w,8,0, ipa_bias.b, bzb,8,0, NN,8,128, 1.f), 1);
        k_points_q <<<NRES,  64, 0, st>>>(qraw,  Rb, tb, qpt);
        k_points_kv<<<NRES, 160, 0, st>>>(kvraw, Rb, tb, kpt, vpt);
        // a0 = q.k^T * sqrt(1/768) per head
        gemm(st,1,0,0, GP(qb,2048,256, kvb,4096,512, nullptr, ab,NRES,NN,
                          NRES,NRES,256, 0.03608439182435161f), 8);
        k_ipa_att<<<dim3(NRES,8), 128, 0, st>>>(ab, bzb, qpt, kpt, head_w, mask);
        // o = a@v per head into cat[:,0:2048]
        gemm(st,0,0,0, GP(ab,NRES,NN, kvb+256,4096,512, nullptr, cat,3456,256,
                          NRES,256,NRES, 1.f), 8);
        // o_pt = a@v_pts per head
        gemm(st,0,0,0, GP(ab,NRES,NN, vpt,288,36, nullptr, optb,288,36,
                          NRES,36,NRES, 1.f), 8);
        // o_pair = a@z, batched over residues, into cat[:,2432:3456]
        gemm(st,0,0,0, GP(ab,NN,NRES, zcur,128,(long)NRES*128, nullptr,
                          cat+2432,128,3456, 8,128,NRES, 1.f), NRES);
        k_optlocal<<<NRES, 96, 0, st>>>(optb, Rb, tb, cat);
        // ipa_out*mask + s -> tmp1 ; LN -> s
        { GemmP p = GP(cat,3456,0, out.w,256,0, out.b, tmp1,256,0, NRES,256,3456, 1.f);
          p.rowscale = mask; p.add = s; p.ldadd = 256; gemm(st,0,0,0,p,1); }
        k_layernorm<<<NRES,128,0,st>>>(tmp1,256, s,256, ipa_ln.g, ipa_ln.b, 256, nullptr,nullptr,1);

        // ---- transformer ----
        k_copycols<<<NRES,128,0,st>>>(s, xb);
        gemm(st,0,0,0, GP(ine,256,0, skip.w,64,0, skip.b, xb+256,320,0, NRES,64,256, 1.f), 1);
        for (int l = 0; l < 2; ++l) {
            gemm(st,0,0,0, GP(xb,320,0, tf_qkv[l].w,960,0, tf_qkv[l].b, qkvb,960,0,
                              NRES,960,320, 1.f), 1);
            gemm(st,1,0,0, GP(qkvb,960,80, qkvb+320,960,80, nullptr, attb,NRES,NN,
                              NRES,NRES,80, 0.11180339887498949f), 4);
            k_softmax_tfmr<<<dim3(NRES,4),128,0,st>>>(attb, mask);
            gemm(st,0,0,0, GP(attb,NRES,NN, qkvb+640,960,80, nullptr, ob,320,80,
                              NRES,80,NRES, 1.f), 4);
            { GemmP p = GP(ob,320,0, tf_o[l].w,320,0, tf_o[l].b, t320,320,0, NRES,320,320, 1.f);
              p.add = xb; p.ldadd = 320; gemm(st,0,0,0,p,1); }
            k_layernorm<<<NRES,128,0,st>>>(t320,320, xb,320, tf_ln1[l].g, tf_ln1[l].b, 320, nullptr,nullptr,1);
            gemm(st,0,1,0, GP(xb,320,0, tf_ff1[l].w,320,0, tf_ff1[l].b, ffb,320,0, NRES,320,320, 1.f), 1);
            { GemmP p = GP(ffb,320,0, tf_ff2[l].w,320,0, tf_ff2[l].b, t320,320,0, NRES,320,320, 1.f);
              p.add = xb; p.ldadd = 320; gemm(st,0,0,0,p,1); }
            k_layernorm<<<NRES,128,0,st>>>(t320,320, xb,320, tf_ln2[l].g, tf_ln2[l].b, 320, nullptr,nullptr,1);
        }
        { GemmP p = GP(xb,320,0, post.w,256,0, post.b, s,256,0, NRES,256,320, 1.f);
          p.add = s; p.ldadd = 256; gemm(st,0,0,0,p,1); }

        // ---- node transition ----
        gemm(st,0,1,0, GP(s,256,0,    l1.w,256,0, l1.b, tmp1,256,0, NRES,256,256, 1.f), 1);
        gemm(st,0,1,0, GP(tmp1,256,0, l2.w,256,0, l2.b, tmp2,256,0, NRES,256,256, 1.f), 1);
        { GemmP p = GP(tmp2,256,0, l3.w,256,0, l3.b, tmp1,256,0, NRES,256,256, 1.f);
          p.add = s; p.ldadd = 256; gemm(st,0,0,0,p,1); }
        k_layernorm<<<NRES,128,0,st>>>(tmp1,256, s,256, nt_ln.g, nt_ln.b, 256, mask,nullptr,1);

        // ---- backbone update + compose ----
        k_maskrows<<<NRES,128,0,st>>>(s, dmsk, tmp2, 256);
        gemm(st,0,0,0, GP(tmp2,256,0, bb.w,6,0, bb.b, updb,6,0, NRES,6,256, 1.f), 1);
        k_compose<<<3,128,0,st>>>(updb, dmsk, Rb, tb);

        // ---- edge transition ----
        if (has_edge) {
            float* zdst = flip ? zB : zA;
            gemm(st,0,0,0, GP(s,256,0, e_init.w,128,0, e_init.b, neb,128,0, NRES,128,256, 1.f), 1);
            gemm(st,0,0,0, GP(neb,128,0, e_tr0.w + 128*384,384,0, nullptr, p1b,384,0, NRES,384,128, 1.f), 1);
            gemm(st,0,0,0, GP(neb,128,0, e_tr0.w + 256*384,384,0, nullptr, q1b,384,0, NRES,384,128, 1.f), 1);
            gemm(st,0,0,0, GP(neb,128,0, e_final.w + 128*128,128,0, nullptr, pfb,128,0, NRES,128,128, 1.f), 1);
            gemm(st,0,0,0, GP(neb,128,0, e_final.w + 256*128,128,0, nullptr, qfb,128,0, NRES,128,128, 1.f), 1);
            for (int c = 0; c < 8; ++c) {
                const int Mc = 48*384;
                const float* zc = zcur + (size_t)c*Mc*128;
                float* ztc = zdst + (size_t)c*Mc*128;
                // x1 = relu(z@W1z + P1[i] + Q1[j] + b1)
                { GemmP p = GP(zc,128,0, e_tr0.w,384,0, e_tr0.b, x1b,384,0, Mc,384,128, 1.f);
                  p.P = p1b + (size_t)c*48*384; p.Q = q1b; p.NJ = 384; p.ldPQ = 384;
                  gemm(st,0,1,1,p,1); }
                // x2 = relu(x1@W2 + b2)
                gemm(st,0,1,0, GP(x1b,384,0, e_tr1.w,384,0, e_tr1.b, x2b,384,0, Mc,384,384, 1.f), 1);
                // zt = x2@Wf + Pf[i] + Qf[j] + bf
                { GemmP p = GP(x2b,384,0, e_final.w,128,0, e_final.b, ztc,128,0, Mc,128,384, 1.f);
                  p.P = pfb + (size_t)c*48*128; p.Q = qfb; p.NJ = 384; p.ldPQ = 128;
                  gemm(st,0,0,1,p,1); }
                // zt += z@Wf[0:128]
                { GemmP p = GP(zc,128,0, e_final.w,128,0, nullptr, ztc,128,0, Mc,128,128, 1.f);
                  p.add = ztc; p.ldadd = 128; gemm(st,0,0,0,p,1); }
            }
            k_layernorm<<<NN,128,0,st>>>(zdst,128, zdst,128, e_ln.g, e_ln.b, 128, mask, mask, 384);
            zcur = zdst; flip = !flip;
        }
    }

    // ---- torsion head + outputs ----
    Lin t_final = RL(); Lin t_l1 = RL(); Lin t_l2 = RL();
    gemm(st,0,1,0, GP(s,256,0, t_l1.w,256,0, t_l1.b, tmp1,256,0, NRES,256,256, 1.f), 1);
    { GemmP p = GP(tmp1,256,0, t_l2.w,256,0, t_l2.b, tmp2,256,0, NRES,256,256, 1.f);
      p.add = s; p.ldadd = 256; gemm(st,0,0,0,p,1); }
    gemm(st,0,0,0, GP(tmp2,256,0, t_final.w,2,0, t_final.b, unb,2,0, NRES,2,256, 1.f), 1);

    float* outp = (float*)d_out;
    k_copyn<<<(98304+255)/256,256,0,st>>>(s, outp, 98304);
    k_out_tail<<<3,128,0,st>>>(unb, tb, Rb, outp);
}
